// TransLayer_40853728920343
// MI455X (gfx1250) — matrix operand order repaired and checked by reading, not compiled, not run on hardware
//
#include <hip/hip_runtime.h>
#include <hip/hip_bf16.h>

// ---------------- types for WMMA ----------------
typedef __bf16 bf16_t;
typedef __attribute__((ext_vector_type(16))) __bf16 v16bf;
typedef __attribute__((ext_vector_type(8)))  __bf16 v8bf;
typedef __attribute__((ext_vector_type(8)))  float  v8f;

union BF16x16 { v16bf v; v8bf h[2]; };

#define DIM 512
#define HEADS 8
#define DH 64
#define SEQ 4096
#define MLM 256
#define LSZ 16
#define KSZ 33

// wave tile: WMT x WNT 16x16 WMMA tiles (32 x 64 C elements per wave)
#define WMT 2
#define WNT 4

// =====================================================================
// Generic bf16 NT GEMM with f32 accumulate and 2x4 register blocking:
//   C[b][m][n] = sum_k A[b][m*K+k] * B[b][n*K+k]  (+bias[n]) (+residual[m*N+n])
// Each wave computes a 32x64 C block via 8 V_WMMA_F32_16X16X32_BF16 per K step.
// A layout per lane (16-bit A 16x32, ISA 7.12.2): lanes 0-15 -> K {0..7,16..23},
// lanes 16-31 -> K {8..15,24..31}; M = lane&15.
// B layout per lane: lanes 0-15 -> K {0..7,16..23}, lanes 16-31 -> K {8..15,24..31}; N = lane&15.
// =====================================================================
__global__ __launch_bounds__(128) void k_gemm_nt(
    const bf16_t* __restrict__ A, const bf16_t* __restrict__ B,
    float* __restrict__ C,
    int M, int N, int K,
    long sA, long sB, long sC,
    const float* __restrict__ bias, const float* __restrict__ residual)
{
  int batch = blockIdx.z;
  A += (long)batch * sA;
  B += (long)batch * sB;
  C += (long)batch * sC;

  int lane = threadIdx.x & 31;
  int wave = threadIdx.x >> 5;
  int n0 = blockIdx.x * (16 * WNT);
  int m0 = (blockIdx.y * 4 + wave) * (16 * WMT);
  if (m0 >= M || n0 >= N) return;  // wave-uniform: EXEC stays all-ones for WMMA

  int kh  = lane >> 4;   // half-wave selector
  int l15 = lane & 15;

  const bf16_t* pa[WMT];
  const bf16_t* pb[WNT];
#pragma unroll
  for (int i = 0; i < WMT; i++) pa[i] = A + (long)(m0 + i * 16 + l15) * K;
#pragma unroll
  for (int j = 0; j < WNT; j++) pb[j] = B + (long)(n0 + j * 16 + l15) * K;

  v8f acc[WMT][WNT] = {};

  for (int k0 = 0; k0 < K; k0 += 32) {
    BF16x16 av[WMT], bv[WNT];
#pragma unroll
    for (int i = 0; i < WMT; i++) {
      av[i].h[0] = *(const v8bf*)(pa[i] + k0 + kh * 8);
      av[i].h[1] = *(const v8bf*)(pa[i] + k0 + 16 + kh * 8);
    }
#pragma unroll
    for (int j = 0; j < WNT; j++) {
      bv[j].h[0] = *(const v8bf*)(pb[j] + k0 + kh * 8);
      bv[j].h[1] = *(const v8bf*)(pb[j] + k0 + 16 + kh * 8);
    }
    if (k0 + 32 < K) {
      __builtin_prefetch(pa[0] + k0 + 96, 0, 0);
      __builtin_prefetch(pb[0] + k0 + 96, 0, 0);
    }
#pragma unroll
    for (int i = 0; i < WMT; i++)
#pragma unroll
      for (int j = 0; j < WNT; j++)
        acc[i][j] = __builtin_amdgcn_wmma_f32_16x16x32_bf16(
            false, av[i].v, false, bv[j].v, (short)0, acc[i][j], false, false);
  }

  // C tile layout (ISA 7.12.2): VGPR r holds M = tile_m + r + 8*(lane>=16),
  // N = tile_n + (lane&15)
#pragma unroll
  for (int j = 0; j < WNT; j++) {
    int cn = n0 + j * 16 + l15;
    float bv2 = bias ? bias[cn] : 0.0f;
#pragma unroll
    for (int i = 0; i < WMT; i++) {
#pragma unroll
      for (int r = 0; r < 8; r++) {
        int cm = m0 + i * 16 + r + kh * 8;
        float val = acc[i][j][r] + bv2;
        if (residual) val += residual[(long)cm * N + cn];
        C[(long)cm * N + cn] = val;
      }
    }
  }
}

// ---------------- LayerNorm (one block per row) ----------------
__global__ void k_layernorm(const float* __restrict__ x,
                            const float* __restrict__ w,
                            const float* __restrict__ b,
                            bf16_t* __restrict__ out, int D)
{
  long row = blockIdx.x;
  const float* xr = x + row * D;
  __shared__ float red[32];
  int wid = threadIdx.x >> 5, ln = threadIdx.x & 31, nw = blockDim.x >> 5;
  float s = 0.f, s2 = 0.f;
  for (int i = threadIdx.x; i < D; i += blockDim.x) {
    float v = xr[i]; s += v; s2 += v * v;
  }
  for (int o = 16; o > 0; o >>= 1) { s += __shfl_xor(s, o, 32); s2 += __shfl_xor(s2, o, 32); }
  if (ln == 0) red[wid] = s;
  __syncthreads();
  s = 0.f; for (int i = 0; i < nw; i++) s += red[i];
  __syncthreads();
  if (ln == 0) red[wid] = s2;
  __syncthreads();
  s2 = 0.f; for (int i = 0; i < nw; i++) s2 += red[i];
  float mean = s / D;
  float var  = s2 / D - mean * mean;
  float inv  = rsqrtf(var + 1e-5f);
  for (int i = threadIdx.x; i < D; i += blockDim.x) {
    float v = (xr[i] - mean) * inv * w[i] + b[i];
    out[row * D + i] = (bf16_t)v;
  }
}

// ---------------- flat f32 -> bf16 cast ----------------
__global__ void k_cast_bf16(const float* __restrict__ in, bf16_t* __restrict__ out, long n)
{
  long i = (long)blockIdx.x * blockDim.x + threadIdx.x;
  if (i < n) out[i] = (bf16_t)in[i];
}

// ---------------- split qkv -> heads, scale q, build v / vT ----------------
__global__ void k_split(const float* __restrict__ qkv,
                        bf16_t* __restrict__ q_b, bf16_t* __restrict__ k_b,
                        float* __restrict__ v_f, bf16_t* __restrict__ vT_b, int N)
{
  long idx = (long)blockIdx.x * blockDim.x + threadIdx.x;  // over N*512
  if (idx >= (long)N * DIM) return;
  int d = idx % DH;
  int h = (idx / DH) % HEADS;
  long n = idx / DIM;
  const float* row = qkv + n * (3 * DIM);
  float qv = row[h * DH + d] * 0.125f;        // DIM_HEAD^-0.5
  float kv = row[DIM + h * DH + d];
  float vv = row[2 * DIM + h * DH + d];
  long hb = (long)h * N * DH;
  q_b[hb + n * DH + d] = (bf16_t)qv;
  k_b[hb + n * DH + d] = (bf16_t)kv;
  v_f[hb + n * DH + d] = vv;
  vT_b[hb + (long)d * N + n] = (bf16_t)vv;    // (DH, N) per head
}

// ---------------- landmark means (ql scaled like q) ----------------
__global__ void k_landmarks(const float* __restrict__ qkv,
                            bf16_t* __restrict__ ql_b, bf16_t* __restrict__ kl_b)
{
  int m = blockIdx.x, h = blockIdx.y, d = threadIdx.x;  // m<256, h<8, d<64
  float sq = 0.f, sk = 0.f;
  for (int j = 0; j < LSZ; j++) {
    const float* row = qkv + (long)(m * LSZ + j) * (3 * DIM) + h * DH + d;
    sq += row[0];
    sk += row[DIM];
  }
  const float invL = 1.0f / LSZ;
  ql_b[((long)h * MLM + m) * DH + d] = (bf16_t)(sq * invL * 0.125f);
  kl_b[((long)h * MLM + m) * DH + d] = (bf16_t)(sk * invL);
}

// ---------------- row softmax (one block per row) ----------------
__global__ void k_softmax(const float* __restrict__ in,
                          float* __restrict__ outf, bf16_t* __restrict__ outb, int len)
{
  long row = blockIdx.x;
  const float* r = in + row * (long)len;
  __shared__ float red[32];
  int wid = threadIdx.x >> 5, ln = threadIdx.x & 31, nw = blockDim.x >> 5;

  float mx = -3.4e38f;
  for (int i = threadIdx.x; i < len; i += blockDim.x) mx = fmaxf(mx, r[i]);
  for (int o = 16; o > 0; o >>= 1) mx = fmaxf(mx, __shfl_xor(mx, o, 32));
  if (ln == 0) red[wid] = mx;
  __syncthreads();
  mx = red[0];
  for (int i = 1; i < nw; i++) mx = fmaxf(mx, red[i]);
  __syncthreads();

  float s = 0.f;
  for (int i = threadIdx.x; i < len; i += blockDim.x) s += __expf(r[i] - mx);
  for (int o = 16; o > 0; o >>= 1) s += __shfl_xor(s, o, 32);
  if (ln == 0) red[wid] = s;
  __syncthreads();
  s = 0.f; for (int i = 0; i < nw; i++) s += red[i];
  float inv = 1.0f / s;

  for (int i = threadIdx.x; i < len; i += blockDim.x) {
    float v = __expf(r[i] - mx) * inv;
    if (outf) outf[row * (long)len + i] = v;
    if (outb) outb[row * (long)len + i] = (bf16_t)v;
  }
}

// ---------------- affine + optional transpose bf16 writer ----------------
// outN[r*C+c] = aN*in[r*C+c] + bN*(r==c) ; outT[c*R+r] = aT*in[r*C+c] + bT*(r==c)
__global__ void k_affine(const float* __restrict__ in, int R, int C,
                         float aN, float bN, bf16_t* __restrict__ outN,
                         float aT, float bT, bf16_t* __restrict__ outT)
{
  long sz = (long)R * C;
  long base = (long)blockIdx.y * sz;
  long idx = (long)blockIdx.x * blockDim.x + threadIdx.x;
  if (idx >= sz) return;
  int rr = idx / C, cc = idx % C;
  float v = in[base + idx];
  float diag = (rr == cc) ? 1.0f : 0.0f;
  if (outN) outN[base + idx] = (bf16_t)(aN * v + bN * diag);
  if (outT) outT[base + (long)cc * R + rr] = (bf16_t)(aT * v + bT * diag);
}

// ---------------- pinv denominator: max row-sum * max col-sum per head ----------------
__global__ void k_denom(const float* __restrict__ a2, float* __restrict__ denom)
{
  int h = blockIdx.x;
  const float* A = a2 + (long)h * MLM * MLM;
  int i = threadIdx.x;  // 256 threads
  float rs = 0.f, cs = 0.f;
  for (int j = 0; j < MLM; j++) { rs += fabsf(A[i * MLM + j]); cs += fabsf(A[j * MLM + i]); }
  __shared__ float shr[MLM], shc[MLM];
  shr[i] = rs; shc[i] = cs;
  __syncthreads();
  for (int o = 128; o > 0; o >>= 1) {
    if (i < o) { shr[i] = fmaxf(shr[i], shr[i + o]); shc[i] = fmaxf(shc[i], shc[i + o]); }
    __syncthreads();
  }
  if (i == 0) denom[h] = shr[0] * shc[0];
}

// ---------------- z0 = a2^T / denom (bf16 normal + transposed) ----------------
__global__ void k_zinit(const float* __restrict__ a2, const float* __restrict__ denom,
                        bf16_t* __restrict__ z_b, bf16_t* __restrict__ zT_b)
{
  int h = blockIdx.y;
  long base = (long)h * MLM * MLM;
  long idx = (long)blockIdx.x * blockDim.x + threadIdx.x;
  int m = idx >> 8, n = idx & 255;
  float inv = 1.0f / denom[h];
  z_b [base + idx] = (bf16_t)(a2[base + (long)n * MLM + m] * inv);
  zT_b[base + idx] = (bf16_t)(a2[base + (long)m * MLM + n] * inv);
}

// ---------------- depthwise conv along sequence (kernel 33) ----------------
__global__ void k_conv(const float* __restrict__ v, const float* __restrict__ w,
                       float* __restrict__ res, int N)
{
  int n = blockIdx.x, h = blockIdx.y, d = threadIdx.x;
  const float* vh = v + (long)h * N * DH;
  const float* wh = w + h * KSZ;
  float s = 0.f;
  for (int j = 0; j < KSZ; j++) {
    int nn = n + j - KSZ / 2;
    if (nn >= 0 && nn < N) s += wh[j] * vh[(long)nn * DH + d];
  }
  res[(long)h * N * DH + (long)n * DH + d] = s;
}

// ---------------- combine heads + conv residual -> (N, INNER) bf16 ----------------
__global__ void k_combine(const float* __restrict__ o, const float* __restrict__ res,
                          bf16_t* __restrict__ oin, int N)
{
  long idx = (long)blockIdx.x * blockDim.x + threadIdx.x;  // over N*512
  if (idx >= (long)N * DIM) return;
  int d = idx % DH;
  int h = (idx / DH) % HEADS;
  long n = idx / DIM;
  long src = (long)h * N * DH + n * DH + d;
  oin[idx] = (bf16_t)(o[src] + res[src]);
}

// =====================================================================
// Host side
// =====================================================================
static inline void gemm(hipStream_t s,
                        const bf16_t* A, const bf16_t* B, float* C,
                        int M, int N, int K,
                        long sA, long sB, long sC, int batch,
                        const float* bias = nullptr, const float* residual = nullptr)
{
  dim3 g(N / (16 * WNT), (M + 4 * 16 * WMT - 1) / (4 * 16 * WMT), batch);
  k_gemm_nt<<<g, 128, 0, s>>>(A, B, C, M, N, K, sA, sB, sC, bias, residual);
}

extern "C" void kernel_launch(void* const* d_in, const int* in_sizes, int n_in,
                              void* d_out, int out_size, void* d_ws, size_t ws_size,
                              hipStream_t stream)
{
  const float* x      = (const float*)d_in[0];  // (1,4096,512)
  const float* norm_w = (const float*)d_in[1];
  const float* norm_b = (const float*)d_in[2];
  const float* qkv_w  = (const float*)d_in[3];  // (1536,512)
  const float* out_w  = (const float*)d_in[4];  // (512,512)
  const float* out_b  = (const float*)d_in[5];
  const float* conv_w = (const float*)d_in[6];  // (8,1,33,1)

  float* y_out    = (float*)d_out;                         // (4096,512)
  float* attn_out = (float*)d_out + (long)SEQ * DIM;       // (8,4096,4096)

  // ---- bump allocator on workspace ----
  char* cur = (char*)d_ws;
  auto alloc = [&](size_t bytes) -> void* {
    void* p = (void*)cur;
    cur += (bytes + 255) & ~(size_t)255;
    return p;
  };
  const long HS  = (long)SEQ * DH;     // per-head seq x 64
  const long HM  = (long)MLM * DH;     // per-head 256 x 64
  const long MM  = (long)MLM * MLM;    // 256 x 256
  const long SM  = (long)SEQ * MLM;    // 4096 x 256

  bf16_t* xn_b    = (bf16_t*)alloc((long)SEQ * DIM * 2);
  bf16_t* qkvw_b  = (bf16_t*)alloc((long)3 * DIM * DIM * 2);
  bf16_t* outw_b  = (bf16_t*)alloc((long)DIM * DIM * 2);
  float*  qkv_f   = (float*) alloc((long)SEQ * 3 * DIM * 4);
  bf16_t* q_b     = (bf16_t*)alloc(HEADS * HS * 2);
  bf16_t* k_b     = (bf16_t*)alloc(HEADS * HS * 2);
  float*  v_f     = (float*) alloc(HEADS * HS * 4);
  bf16_t* vT_b    = (bf16_t*)alloc(HEADS * HS * 2);
  bf16_t* ql_b    = (bf16_t*)alloc(HEADS * HM * 2);
  bf16_t* kl_b    = (bf16_t*)alloc(HEADS * HM * 2);
  float*  sim1_f  = (float*) alloc(HEADS * SM * 4);
  float*  sim2_f  = (float*) alloc(HEADS * MM * 4);
  float*  sim3_f  = (float*) alloc(HEADS * SM * 4);
  bf16_t* a1_b    = (bf16_t*)alloc(HEADS * SM * 2);
  float*  a2_f    = (float*) alloc(HEADS * MM * 4);
  bf16_t* a2_b    = (bf16_t*)alloc(HEADS * MM * 2);
  float*  a3_f    = (float*) alloc(HEADS * SM * 4);
  bf16_t* a3_b    = (bf16_t*)alloc(HEADS * SM * 2);
  bf16_t* a3T_b   = (bf16_t*)alloc(HEADS * SM * 2);
  float*  denom_f = (float*) alloc(HEADS * 4);
  bf16_t* z_b     = (bf16_t*)alloc(HEADS * MM * 2);
  bf16_t* zT_b    = (bf16_t*)alloc(HEADS * MM * 2);
  bf16_t* xz_b    = (bf16_t*)alloc(HEADS * MM * 2);
  bf16_t* w1T_b   = (bf16_t*)alloc(HEADS * MM * 2);
  bf16_t* w2T_b   = (bf16_t*)alloc(HEADS * MM * 2);
  bf16_t* w3T_b   = (bf16_t*)alloc(HEADS * MM * 2);
  float*  P_f     = (float*) alloc(HEADS * MM * 4);
  float*  t_f     = (float*) alloc(HEADS * HM * 4);
  bf16_t* tT_b    = (bf16_t*)alloc(HEADS * HM * 2);
  float*  s_f     = (float*) alloc(HEADS * HM * 4);
  bf16_t* sT_b    = (bf16_t*)alloc(HEADS * HM * 2);
  float*  o_f     = (float*) alloc(HEADS * HS * 4);
  float*  r_f     = (float*) alloc(HEADS * SM * 4);
  bf16_t* rT_b    = (bf16_t*)alloc(HEADS * SM * 2);
  float*  res_f   = (float*) alloc(HEADS * HS * 4);
  bf16_t* oin_b   = (bf16_t*)alloc((long)SEQ * DIM * 2);

  // ---- 1. weight casts + LayerNorm ----
  {
    long n = (long)3 * DIM * DIM;
    k_cast_bf16<<<(n + 255) / 256, 256, 0, stream>>>(qkv_w, qkvw_b, n);
    n = (long)DIM * DIM;
    k_cast_bf16<<<(n + 255) / 256, 256, 0, stream>>>(out_w, outw_b, n);
  }
  k_layernorm<<<SEQ, 128, 0, stream>>>(x, norm_w, norm_b, xn_b, DIM);

  // ---- 2. QKV projection: (4096,512) x (1536,512)^T ----
  gemm(stream, xn_b, qkvw_b, qkv_f, SEQ, 3 * DIM, DIM, 0, 0, 0, 1);

  // ---- 3. split to heads / landmarks ----
  {
    long n = (long)SEQ * DIM;
    k_split<<<(n + 255) / 256, 256, 0, stream>>>(qkv_f, q_b, k_b, v_f, vT_b, SEQ);
  }
  k_landmarks<<<dim3(MLM, HEADS), DH, 0, stream>>>(qkv_f, ql_b, kl_b);

  // ---- 4. similarity GEMMs (batched over heads) ----
  gemm(stream, q_b,  kl_b, sim1_f, SEQ, MLM, DH, HS, HM, SM, HEADS);
  gemm(stream, ql_b, kl_b, sim2_f, MLM, MLM, DH, HM, HM, MM, HEADS);
  gemm(stream, ql_b, k_b,  sim3_f, MLM, SEQ, DH, HM, HS, SM, HEADS);

  // ---- 5. softmaxes ----
  k_softmax<<<HEADS * SEQ, 128, 0, stream>>>(sim1_f, nullptr, a1_b, MLM);
  k_softmax<<<HEADS * MLM, 128, 0, stream>>>(sim2_f, a2_f, a2_b, MLM);
  k_softmax<<<HEADS * MLM, 256, 0, stream>>>(sim3_f, a3_f, a3_b, SEQ);
  // a3^T bf16 (for a2inv @ a3 as NT)
  k_affine<<<dim3((SM + 255) / 256, HEADS), 256, 0, stream>>>(
      a3_f, MLM, SEQ, 0.f, 0.f, nullptr, 1.f, 0.f, a3T_b);

  // ---- 6. Moore-Penrose pinv (Newton-Schulz, 6 iters) ----
  k_denom<<<HEADS, MLM, 0, stream>>>(a2_f, denom_f);
  k_zinit<<<dim3((MM + 255) / 256, HEADS), 256, 0, stream>>>(a2_f, denom_f, z_b, zT_b);

  dim3 gAff((MM + 255) / 256, HEADS);
  for (int it = 0; it < 6; it++) {
    // xz = a2 @ z
    gemm(stream, a2_b, zT_b, P_f, MLM, MLM, MLM, MM, MM, MM, HEADS);
    k_affine<<<gAff, 256, 0, stream>>>(P_f, MLM, MLM, 1.f, 0.f, xz_b, -1.f, 7.f, w1T_b);
    // w2 = 15I - xz @ w1
    gemm(stream, xz_b, w1T_b, P_f, MLM, MLM, MLM, MM, MM, MM, HEADS);
    k_affine<<<gAff, 256, 0, stream>>>(P_f, MLM, MLM, 0.f, 0.f, nullptr, -1.f, 15.f, w2T_b);
    // w3 = 13I - xz @ w2
    gemm(stream, xz_b, w2T_b, P_f, MLM, MLM, MLM, MM, MM, MM, HEADS);
    k_affine<<<gAff, 256, 0, stream>>>(P_f, MLM, MLM, 0.f, 0.f, nullptr, -1.f, 13.f, w3T_b);
    // z = 0.25 * z @ w3
    gemm(stream, z_b, w3T_b, P_f, MLM, MLM, MLM, MM, MM, MM, HEADS);
    k_affine<<<gAff, 256, 0, stream>>>(P_f, MLM, MLM, 0.25f, 0.f, z_b, 0.25f, 0.f, zT_b);
  }

  // ---- 7. out = a1 @ a2inv @ (a3 @ v) ----
  gemm(stream, a3_b, vT_b, t_f, MLM, DH, SEQ, SM, HS, HM, HEADS);           // t = a3 @ v
  k_affine<<<dim3((HM + 255) / 256, HEADS), 256, 0, stream>>>(
      t_f, MLM, DH, 0.f, 0.f, nullptr, 1.f, 0.f, tT_b);
  gemm(stream, z_b, tT_b, s_f, MLM, DH, MLM, MM, HM, HM, HEADS);            // s = a2inv @ t
  k_affine<<<dim3((HM + 255) / 256, HEADS), 256, 0, stream>>>(
      s_f, MLM, DH, 0.f, 0.f, nullptr, 1.f, 0.f, sT_b);
  gemm(stream, a1_b, sT_b, o_f, SEQ, DH, MLM, SM, HM, HS, HEADS);           // o = a1 @ s

  // ---- 8. attn = a1 @ (a2inv @ a3)  (written straight to output) ----
  gemm(stream, z_b, a3T_b, r_f, MLM, SEQ, MLM, MM, SM, SM, HEADS);          // r = a2inv @ a3
  k_affine<<<dim3((SM + 255) / 256, HEADS), 256, 0, stream>>>(
      r_f, MLM, SEQ, 0.f, 0.f, nullptr, 1.f, 0.f, rT_b);
  gemm(stream, a1_b, rT_b, attn_out, SEQ, SEQ, MLM, SM, SM, (long)SEQ * SEQ, HEADS);

  // ---- 9. depthwise conv residual + combine ----
  k_conv<<<dim3(SEQ, HEADS), DH, 0, stream>>>(v_f, conv_w, res_f, SEQ);
  {
    long n = (long)SEQ * DIM;
    k_combine<<<(n + 255) / 256, 256, 0, stream>>>(o_f, res_f, oin_b, SEQ);
  }

  // ---- 10. output projection + bias + residual(x), written to d_out ----
  gemm(stream, oin_b, outw_b, y_out, SEQ, DIM, DIM, 0, 0, 0, 1, out_b, x);
}
